// FluidInteractionBlock_55173149884914
// MI455X (gfx1250) — compile-verified, matrix operand order checked by reading, not run on hardware
//
#include <hip/hip_runtime.h>
#include <hip/hip_bf16.h>

typedef __attribute__((ext_vector_type(16))) _Float16 v16h;
typedef __attribute__((ext_vector_type(8)))  _Float16 v8h;
typedef __attribute__((ext_vector_type(8)))  float    v8f;

#define HDIM 128
#define NT 8   // 8 column tiles of 16 = 128 cols; also 8 row tiles of 16 = 128 rows/block

// ---------------------------------------------------------------------------
// helpers
// ---------------------------------------------------------------------------
__device__ __forceinline__ void reduce16_8(float v[8]) {
#pragma unroll
  for (int m = 1; m < 16; m <<= 1) {
#pragma unroll
    for (int r = 0; r < 8; ++r) v[r] += __shfl_xor(v[r], m, 32);
  }
}

// A-fragment layout in LDS per K-chunk: [tile(8)][lane(32)][16 halves], 4096 halves/chunk.
// B-fragment layout in global: frag (chunk,ntile) = 32 lanes x 16 halves contiguous (512 halves).
__device__ __forceinline__ void wmma_gemm(const _Float16* sA, const _Float16* Bw,
                                          int nch, int wv, int lane, v8f* acc) {
  for (int c = 0; c < nch; ++c) {
    v16h b = ((const v16h*)Bw)[(c * NT + wv) * 32 + lane];
    const _Float16* ab = sA + c * 4096;
#pragma unroll
    for (int t = 0; t < NT; ++t) {
      v16h a = *(const v16h*)(ab + (t * 32 + lane) * 16);
      acc[t] = __builtin_amdgcn_wmma_f32_16x16x32_f16(
          false, a, false, b, (short)0, acc[t], false, false);
    }
  }
}

// Fused dual-GEMM sharing A fragments (one LDS read feeds two WMMAs).
__device__ __forceinline__ void wmma_gemm2(const _Float16* sA,
                                           const _Float16* Bw0, const _Float16* Bw1,
                                           int nch, int wv, int lane,
                                           v8f* acc0, v8f* acc1) {
  for (int c = 0; c < nch; ++c) {
    v16h b0 = ((const v16h*)Bw0)[(c * NT + wv) * 32 + lane];
    v16h b1 = ((const v16h*)Bw1)[(c * NT + wv) * 32 + lane];
    const _Float16* ab = sA + c * 4096;
#pragma unroll
    for (int t = 0; t < NT; ++t) {
      v16h a = *(const v16h*)(ab + (t * 32 + lane) * 16);
      acc0[t] = __builtin_amdgcn_wmma_f32_16x16x32_f16(
          false, a, false, b0, (short)0, acc0[t], false, false);
      acc1[t] = __builtin_amdgcn_wmma_f32_16x16x32_f16(
          false, a, false, b1, (short)0, acc1[t], false, false);
    }
  }
}

// Scatter a 16x16 C-tile (bias+ReLU applied) into an A-fragment-layout LDS buffer.
// C layout: lane holds col n = wv*16+(lane&15); VGPR r holds row r + 8*(lane>=16).
__device__ __forceinline__ void store_htile_relu(_Float16* hb, int t, int lane, int col,
                                                 const v8f& a, float bias) {
  const int hiL = lane >> 4;
  const int kk = col & 31, c2 = col >> 5;
  const int hi2 = (kk >> 3) & 1;
  const int j2 = (kk & 7) + ((kk & 16) ? 8 : 0);
#pragma unroll
  for (int r = 0; r < 8; ++r) {
    float v = fmaxf(a[r] + bias, 0.0f);
    int lA = (r + 8 * hiL) + (hi2 << 4);
    hb[c2 * 4096 + (t * 32 + lA) * 16 + j2] = (_Float16)v;
  }
}

// ---------------------------------------------------------------------------
// weight swizzle: row-major [K][128] f32 -> per-fragment f16 layout
// ---------------------------------------------------------------------------
__global__ void swizzle_weights(const float* __restrict__ w, _Float16* __restrict__ out,
                                int nch) {
  int i = blockIdx.x * 256 + threadIdx.x;
  int total = nch * NT * 512;
  if (i >= total) return;
  int j = i & 15, l = (i >> 4) & 31, f = i >> 9;
  int nt = f & 7, c = f >> 3;
  int K = c * 32 + ((l >= 16) ? 8 : 0) + (j & 7) + ((j >= 8) ? 16 : 0);
  int Ncol = nt * 16 + (l & 15);
  out[i] = (_Float16)w[(size_t)K * HDIM + Ncol];
}

__global__ void zero_f32(float* __restrict__ p, long long n) {
  long long i = (long long)blockIdx.x * 256 + threadIdx.x;
  if (i < n) p[i] = 0.0f;
}

// ---------------------------------------------------------------------------
// fused edge kernel: cat -> gate MLP + edge MLP + LN -> e_out, atomic aggr
// LDS: catA (12 chunks, 96KB); h0/h1 alias onto catA after it is consumed.
// ---------------------------------------------------------------------------
__global__ __launch_bounds__(256) void edge_kernel(
    const float* __restrict__ x, const long long* __restrict__ eidx,
    const float* __restrict__ eattr,
    const _Float16* __restrict__ ew0h, const float* __restrict__ eb0,
    const _Float16* __restrict__ ew1h, const float* __restrict__ eb1,
    const _Float16* __restrict__ ew2h, const float* __restrict__ eb2,
    const float* __restrict__ elng, const float* __restrict__ elnb,
    const _Float16* __restrict__ gw0h, const float* __restrict__ gb0,
    const float* __restrict__ gw1, const float* __restrict__ gb1,
    float* __restrict__ aggr, float* __restrict__ eout, int E_) {
  extern __shared__ char smem[];
  _Float16* catA = (_Float16*)smem;                 // 12 chunks * 4096 halves = 98304 B
  _Float16* h0A  = (_Float16*)smem;                 // aliases catA chunks 0-3 (after sync)
  _Float16* h1A  = (_Float16*)(smem + 32768);       // aliases catA chunks 4-7
  float* sGate   = (float*)(smem + 98304);          // 128 f32
  float* sSum    = (float*)(smem + 98816);
  float* sSq     = (float*)(smem + 99328);
  int*   sSrc    = (int*)(smem + 99840);
  int*   sDst    = (int*)(smem + 100352);           // total 100864 B

  const int tid = threadIdx.x;
  const int blockStart = blockIdx.x * 128;

  if (tid < 128) {
    int rG = blockStart + tid;
    sSrc[tid]  = (rG < E_) ? (int)eidx[rG] : 0;
    sDst[tid]  = (rG < E_) ? (int)eidx[E_ + rG] : 0;
    sGate[tid] = gb1[0];
    sSum[tid]  = 0.0f;
    sSq[tid]   = 0.0f;
  }
  __syncthreads();

  // ---- cooperative stage of cat = [x[dst] | x[src] | edge_attr] into A-frag layout
  {
    int t = tid >> 5, l = tid & 31;
    int m = l & 15, hb = (l >> 4) << 3;
    int rowL = t * 16 + m;
    int rG = blockStart + rowL;
    bool valid = rG < E_;
    int iD = sDst[rowL], iS = sSrc[rowL];
    for (int c = 0; c < 12; ++c) {
      int colb = c * 32 + hb;
      const float* sp;
      if (c < 4)      sp = x + (size_t)iD * HDIM + colb;
      else if (c < 8) sp = x + (size_t)iS * HDIM + (colb - 128);
      else            sp = eattr + (size_t)rG * HDIM + (colb - 256);
      float4 r0 = {0,0,0,0}, r1 = {0,0,0,0}, r2 = {0,0,0,0}, r3 = {0,0,0,0};
      if (valid) {
        r0 = *(const float4*)(sp + 0);
        r1 = *(const float4*)(sp + 4);
        r2 = *(const float4*)(sp + 16);
        r3 = *(const float4*)(sp + 20);
      }
      _Float16* d = catA + c * 4096 + (t * 32 + l) * 16;
      v8h lo = {(_Float16)r0.x, (_Float16)r0.y, (_Float16)r0.z, (_Float16)r0.w,
                (_Float16)r1.x, (_Float16)r1.y, (_Float16)r1.z, (_Float16)r1.w};
      v8h hi = {(_Float16)r2.x, (_Float16)r2.y, (_Float16)r2.z, (_Float16)r2.w,
                (_Float16)r3.x, (_Float16)r3.y, (_Float16)r3.z, (_Float16)r3.w};
      *(v8h*)d = lo;
      *(v8h*)(d + 8) = hi;
    }
  }
  __syncthreads();

  const int wv = tid >> 5, lane = tid & 31;
  const int n15 = lane & 15, hiL = lane >> 4;
  const int col = wv * 16 + n15;
  v8f zero = {};
  v8f acc[NT], accg[NT];

  // ---- fused: gate hidden GEMM + edge layer0 GEMM share cat A-fragments (K=384)
  for (int t = 0; t < NT; ++t) { acc[t] = zero; accg[t] = zero; }
  wmma_gemm2(catA, gw0h, ew0h, 12, wv, lane, accg, acc);

  // gate: relu(.)@g_w1, row-reduced into sGate
  {
    float gb = gb0[col], gw = gw1[col];
    for (int t = 0; t < NT; ++t) {
      float p[8];
#pragma unroll
      for (int r = 0; r < 8; ++r) p[r] = fmaxf(accg[t][r] + gb, 0.0f) * gw;
      reduce16_8(p);
      if (n15 == 0) {
#pragma unroll
        for (int r = 0; r < 8; ++r) atomicAdd(&sGate[t * 16 + r + 8 * hiL], p[r]);
      }
    }
  }
  __syncthreads();   // all waves done reading catA + gate adds done

  // ---- store h0 (aliases catA chunks 0-3)
  {
    float b = eb0[col];
    for (int t = 0; t < NT; ++t) store_htile_relu(h0A, t, lane, col, acc[t], b);
  }
  __syncthreads();

  // ---- edge layer 1 (K=128): read h0A, write h1A (disjoint region)
  for (int t = 0; t < NT; ++t) acc[t] = zero;
  wmma_gemm(h0A, ew1h, 4, wv, lane, acc);
  {
    float b = eb1[col];
    for (int t = 0; t < NT; ++t) store_htile_relu(h1A, t, lane, col, acc[t], b);
  }
  __syncthreads();

  // ---- edge layer 2 (K=128), keep in regs; LayerNorm stats
  for (int t = 0; t < NT; ++t) acc[t] = zero;
  wmma_gemm(h1A, ew2h, 4, wv, lane, acc);
  const float b2 = eb2[col];
  for (int t = 0; t < NT; ++t) {
    float s1[8], s2[8];
#pragma unroll
    for (int r = 0; r < 8; ++r) { float v = acc[t][r] + b2; s1[r] = v; s2[r] = v * v; }
    reduce16_8(s1);
    reduce16_8(s2);
    if (n15 == 0) {
#pragma unroll
      for (int r = 0; r < 8; ++r) {
        int rowL = t * 16 + r + 8 * hiL;
        atomicAdd(&sSum[rowL], s1[r]);
        atomicAdd(&sSq[rowL],  s2[r]);
      }
    }
  }
  __syncthreads();

  // ---- finalize: LN, e_out = edge_attr + e_new, aggr += gate * e_new
  const float lg = elng[col], lb = elnb[col];
  for (int t = 0; t < NT; ++t) {
#pragma unroll
    for (int r = 0; r < 8; ++r) {
      int rowL = t * 16 + r + 8 * hiL;
      int rG = blockStart + rowL;
      if (rG < E_) {
        float mu   = sSum[rowL] * (1.0f / 128.0f);
        float var  = sSq[rowL] * (1.0f / 128.0f) - mu * mu;
        float rstd = rsqrtf(var + 1e-5f);
        float v  = acc[t][r] + b2;
        float en = (v - mu) * rstd * lg + lb;
        size_t oi = (size_t)rG * HDIM + col;
        eout[oi] = eattr[oi] + en;
        float z = sGate[rowL];
        float g = 1.0f / (1.0f + __expf(-z));
        atomicAdd(&aggr[(size_t)sDst[rowL] * HDIM + col], g * en);
      }
    }
  }
}

// ---------------------------------------------------------------------------
// fused node kernel: [x | aggr] -> node MLP + LN + residual
// LDS: catA (8 chunks, 64KB); h0/h1 alias onto catA after it is consumed.
// ---------------------------------------------------------------------------
__global__ __launch_bounds__(256) void node_kernel(
    const float* __restrict__ x, const float* __restrict__ aggr,
    const _Float16* __restrict__ nw0h, const float* __restrict__ nb0,
    const _Float16* __restrict__ nw1h, const float* __restrict__ nb1,
    const _Float16* __restrict__ nw2h, const float* __restrict__ nb2,
    const float* __restrict__ nlng, const float* __restrict__ nlnb,
    float* __restrict__ xout, int N_) {
  extern __shared__ char smem[];
  _Float16* catA = (_Float16*)smem;             // 8 chunks = 65536 B
  _Float16* h0A  = (_Float16*)smem;             // aliases catA chunks 0-3
  _Float16* h1A  = (_Float16*)(smem + 32768);   // aliases catA chunks 4-7
  float* sSum    = (float*)(smem + 65536);
  float* sSq     = (float*)(smem + 66048);      // total 66560 B

  const int tid = threadIdx.x;
  const int blockStart = blockIdx.x * 128;

  if (tid < 128) { sSum[tid] = 0.0f; sSq[tid] = 0.0f; }

  // ---- stage cat = [x | aggr]
  {
    int t = tid >> 5, l = tid & 31;
    int m = l & 15, hb = (l >> 4) << 3;
    int rowL = t * 16 + m;
    int rG = blockStart + rowL;
    bool valid = rG < N_;
    for (int c = 0; c < 8; ++c) {
      int colb = c * 32 + hb;
      const float* sp = (c < 4) ? (x + (size_t)rG * HDIM + colb)
                                : (aggr + (size_t)rG * HDIM + (colb - 128));
      float4 r0 = {0,0,0,0}, r1 = {0,0,0,0}, r2 = {0,0,0,0}, r3 = {0,0,0,0};
      if (valid) {
        r0 = *(const float4*)(sp + 0);
        r1 = *(const float4*)(sp + 4);
        r2 = *(const float4*)(sp + 16);
        r3 = *(const float4*)(sp + 20);
      }
      _Float16* d = catA + c * 4096 + (t * 32 + l) * 16;
      v8h lo = {(_Float16)r0.x, (_Float16)r0.y, (_Float16)r0.z, (_Float16)r0.w,
                (_Float16)r1.x, (_Float16)r1.y, (_Float16)r1.z, (_Float16)r1.w};
      v8h hi = {(_Float16)r2.x, (_Float16)r2.y, (_Float16)r2.z, (_Float16)r2.w,
                (_Float16)r3.x, (_Float16)r3.y, (_Float16)r3.z, (_Float16)r3.w};
      *(v8h*)d = lo;
      *(v8h*)(d + 8) = hi;
    }
  }
  __syncthreads();

  const int wv = tid >> 5, lane = tid & 31;
  const int n15 = lane & 15, hiL = lane >> 4;
  const int col = wv * 16 + n15;
  v8f zero = {};
  v8f acc[NT];

  // ---- node layer 0 (K=256)
  for (int t = 0; t < NT; ++t) acc[t] = zero;
  wmma_gemm(catA, nw0h, 8, wv, lane, acc);
  __syncthreads();   // all waves done reading catA before h0 aliases it
  {
    float b = nb0[col];
    for (int t = 0; t < NT; ++t) store_htile_relu(h0A, t, lane, col, acc[t], b);
  }
  __syncthreads();

  // ---- node layer 1 (K=128)
  for (int t = 0; t < NT; ++t) acc[t] = zero;
  wmma_gemm(h0A, nw1h, 4, wv, lane, acc);
  {
    float b = nb1[col];
    for (int t = 0; t < NT; ++t) store_htile_relu(h1A, t, lane, col, acc[t], b);
  }
  __syncthreads();

  // ---- node layer 2 (K=128) + LayerNorm stats
  for (int t = 0; t < NT; ++t) acc[t] = zero;
  wmma_gemm(h1A, nw2h, 4, wv, lane, acc);
  const float b2 = nb2[col];
  for (int t = 0; t < NT; ++t) {
    float s1[8], s2[8];
#pragma unroll
    for (int r = 0; r < 8; ++r) { float v = acc[t][r] + b2; s1[r] = v; s2[r] = v * v; }
    reduce16_8(s1);
    reduce16_8(s2);
    if (n15 == 0) {
#pragma unroll
      for (int r = 0; r < 8; ++r) {
        int rowL = t * 16 + r + 8 * hiL;
        atomicAdd(&sSum[rowL], s1[r]);
        atomicAdd(&sSq[rowL],  s2[r]);
      }
    }
  }
  __syncthreads();

  const float lg = nlng[col], lb = nlnb[col];
  for (int t = 0; t < NT; ++t) {
#pragma unroll
    for (int r = 0; r < 8; ++r) {
      int rowL = t * 16 + r + 8 * hiL;
      int rG = blockStart + rowL;
      if (rG < N_) {
        float mu   = sSum[rowL] * (1.0f / 128.0f);
        float var  = sSq[rowL] * (1.0f / 128.0f) - mu * mu;
        float rstd = rsqrtf(var + 1e-5f);
        float v  = acc[t][r] + b2;
        float nn = (v - mu) * rstd * lg + lb;
        size_t oi = (size_t)rG * HDIM + col;
        xout[oi] = x[oi] + nn;
      }
    }
  }
}

// ---------------------------------------------------------------------------
// launcher
// ---------------------------------------------------------------------------
extern "C" void kernel_launch(void* const* d_in, const int* in_sizes, int n_in,
                              void* d_out, int out_size, void* d_ws, size_t ws_size,
                              hipStream_t stream) {
  (void)n_in; (void)out_size; (void)ws_size;
  const int NN = in_sizes[0] / HDIM;   // 50000
  const int EE = in_sizes[1] / 2;      // 500000

  const float*     x     = (const float*)d_in[0];
  const long long* eidx  = (const long long*)d_in[1];
  const float*     eattr = (const float*)d_in[2];
  const float* e_w0 = (const float*)d_in[3];  const float* e_b0 = (const float*)d_in[4];
  const float* e_w1 = (const float*)d_in[5];  const float* e_b1 = (const float*)d_in[6];
  const float* e_w2 = (const float*)d_in[7];  const float* e_b2 = (const float*)d_in[8];
  const float* elng = (const float*)d_in[9];  const float* elnb = (const float*)d_in[10];
  const float* n_w0 = (const float*)d_in[11]; const float* n_b0 = (const float*)d_in[12];
  const float* n_w1 = (const float*)d_in[13]; const float* n_b1 = (const float*)d_in[14];
  const float* n_w2 = (const float*)d_in[15]; const float* n_b2 = (const float*)d_in[16];
  const float* nlng = (const float*)d_in[17]; const float* nlnb = (const float*)d_in[18];
  const float* g_w0 = (const float*)d_in[19]; const float* g_b0 = (const float*)d_in[20];
  const float* g_w1 = (const float*)d_in[21]; const float* g_b1 = (const float*)d_in[22];

  // workspace carve: [aggr f32 N*H][swizzled f16 weights]
  float* aggr = (float*)d_ws;
  size_t aggrBytes = ((size_t)NN * HDIM * 4 + 255) & ~(size_t)255;
  _Float16* wh = (_Float16*)((char*)d_ws + aggrBytes);
  _Float16* ew0h = wh;               // 12*8*512 = 49152 halves
  _Float16* ew1h = wh + 49152;       // 16384
  _Float16* ew2h = wh + 65536;       // 16384
  _Float16* gw0h = wh + 81920;       // 49152
  _Float16* nw0h = wh + 131072;      // 32768
  _Float16* nw1h = wh + 163840;      // 16384
  _Float16* nw2h = wh + 180224;      // 16384

  // 1) swizzle weights to WMMA B-fragment layout (idempotent)
  swizzle_weights<<<(12 * NT * 512 + 255) / 256, 256, 0, stream>>>(e_w0, ew0h, 12);
  swizzle_weights<<<( 4 * NT * 512 + 255) / 256, 256, 0, stream>>>(e_w1, ew1h, 4);
  swizzle_weights<<<( 4 * NT * 512 + 255) / 256, 256, 0, stream>>>(e_w2, ew2h, 4);
  swizzle_weights<<<(12 * NT * 512 + 255) / 256, 256, 0, stream>>>(g_w0, gw0h, 12);
  swizzle_weights<<<( 8 * NT * 512 + 255) / 256, 256, 0, stream>>>(n_w0, nw0h, 8);
  swizzle_weights<<<( 4 * NT * 512 + 255) / 256, 256, 0, stream>>>(n_w1, nw1h, 4);
  swizzle_weights<<<( 4 * NT * 512 + 255) / 256, 256, 0, stream>>>(n_w2, nw2h, 4);

  // 2) zero aggregation buffer
  long long aggrN = (long long)NN * HDIM;
  zero_f32<<<(int)((aggrN + 255) / 256), 256, 0, stream>>>(aggr, aggrN);

  float* outx = (float*)d_out;
  float* oute = outx + (size_t)NN * HDIM;

  // 3) fused edge pass
  edge_kernel<<<(EE + 127) / 128, 256, 100864, stream>>>(
      x, eidx, eattr, ew0h, e_b0, ew1h, e_b1, ew2h, e_b2, elng, elnb,
      gw0h, g_b0, g_w1, g_b1, aggr, oute, EE);

  // 4) fused node pass
  node_kernel<<<(NN + 127) / 128, 256, 66560, stream>>>(
      x, aggr, nw0h, n_b0, nw1h, n_b1, nw2h, n_b2, nlng, nlnb, outx, NN);
}
